// Attention_9019431321978
// MI455X (gfx1250) — matrix operand order repaired and checked by reading, not compiled, not run on hardware
//
#include <hip/hip_runtime.h>

typedef __bf16 bf16_t;
typedef __attribute__((ext_vector_type(16))) __bf16 v16bf;
typedef __attribute__((ext_vector_type(8)))  __bf16 v8bf;
typedef __attribute__((ext_vector_type(8)))  float  v8f;

#define WMMA_BF16(a, b, c) \
  __builtin_amdgcn_wmma_f32_16x16x32_bf16(false, (a), false, (b), (short)0, (c), false, false)

union BF16x16 { v16bf v; v8bf h[2]; };

__device__ __forceinline__ v8f vzero8() {
  v8f z; for (int i = 0; i < 8; ++i) z[i] = 0.0f; return z;
}

// A-fragment: 16x32 bf16, M = lane&15.
// lane<16: elems 0-7 = K 0..7, elems 8-15 = K 16..23
// lane>=16: elems 0-7 = K 8..15, elems 8-15 = K 24..31
__device__ __forceinline__ v16bf load_a_frag(const bf16_t* base, int ld, int row0, int k0, int lane) {
  int m  = lane & 15;
  int hf = (lane >> 4) & 1;
  const bf16_t* p = base + (size_t)(row0 + m) * ld + k0;
  BF16x16 r;
  r.h[0] = *(const v8bf*)(p + hf * 8);
  r.h[1] = *(const v8bf*)(p + 16 + hf * 8);
  return r.v;
}

// B-fragment: 32x16 bf16, N = lane&15, K rows (lane>=16 ? 8..15,24..31 : 0..7,16..23).
// Column n of B is contiguous in memory (row n of a (N,K) row-major array).
__device__ __forceinline__ v16bf load_b_frag(const bf16_t* base, size_t ld, int n0, int k0, int lane) {
  int n  = lane & 15;
  int kb = (lane >> 4) * 8;
  const bf16_t* p = base + (size_t)(n0 + n) * ld + (size_t)k0 + kb;
  BF16x16 r;
  r.h[0] = *(const v8bf*)p;
  r.h[1] = *(const v8bf*)(p + 16);
  return r.v;
}

// ---------------------------------------------------------------------------
// OUT[M,N] (f32) = A[M,K] (bf16, row-major) @ W[N,K]^T (bf16, row-major)
// Block: 256 threads = 8 waves as 2(M) x 4(N); wave tile 64x64 = 4x4 WMMA.
// Block tile: 128(M) x 256(N). A tile staged in LDS via async copies,
// double-buffered (ASYNCcnt pipelining); B direct from global (L2-resident).
// ---------------------------------------------------------------------------
__global__ __launch_bounds__(256)
void gemm_xt_bf16_f32(const bf16_t* __restrict__ A, const bf16_t* __restrict__ W,
                      float* __restrict__ OUT, int M, int N, int K) {
  constexpr int APAD = 40;                             // row stride (elems), 80B rows
  __shared__ __align__(16) bf16_t aTile[2][128 * APAD];
  int tid  = threadIdx.x;
  int lane = tid & 31;
  int wave = tid >> 5;
  int mg = wave >> 2;                                  // 0..1
  int ng = wave & 3;                                   // 0..3
  int rowBlk = blockIdx.y * 128;
  int col0   = blockIdx.x * 256 + ng * 64;
  int nk = K >> 5;

  // Each thread async-copies two 16B chunks of the 128x32 A tile into LDS.
  auto issueTile = [&](int kt, int buf) {
    int k0 = kt << 5;
#pragma unroll
    for (int ci = 0; ci < 2; ++ci) {
      int ch = tid + ci * 256;                         // 0..511
      int r  = ch >> 2;                                // row 0..127
      int c  = ch & 3;                                 // 8-elem chunk 0..3
      const bf16_t* g = A + (size_t)(rowBlk + r) * K + k0 + c * 8;
      unsigned lo = (unsigned)(size_t)&aTile[buf][r * APAD + c * 8];
      asm volatile("global_load_async_to_lds_b128 %0, %1, off"
                   :: "v"(lo), "v"(g) : "memory");
    }
  };

  issueTile(0, 0);

  v8f acc[4][4];
  for (int i = 0; i < 4; ++i)
    for (int j = 0; j < 4; ++j) acc[i][j] = vzero8();

  int mloc = lane & 15;
  int hf   = (lane >> 4) & 1;

  for (int kt = 0; kt < nk; ++kt) {
    int buf = kt & 1;
    if (kt + 1 < nk) {
      issueTile(kt + 1, buf ^ 1);                      // prefetch next tile
      asm volatile("s_wait_asynccnt 0x2" ::: "memory"); // current tile retired
    } else {
      asm volatile("s_wait_asynccnt 0x0" ::: "memory");
    }
    __syncthreads();                                   // whole tile visible

    int k0 = kt << 5;
    v16bf af[4];
#pragma unroll
    for (int i = 0; i < 4; ++i) {
      const bf16_t* lp = &aTile[buf][(mg * 64 + i * 16 + mloc) * APAD];
      BF16x16 r;
      r.h[0] = *(const v8bf*)(lp + hf * 8);
      r.h[1] = *(const v8bf*)(lp + 16 + hf * 8);
      af[i] = r.v;
    }
#pragma unroll
    for (int j = 0; j < 4; ++j) {
      v16bf bfj = load_b_frag(W, K, col0 + j * 16, k0, lane);
#pragma unroll
      for (int i = 0; i < 4; ++i)
        acc[i][j] = WMMA_BF16(af[i], bfj, acc[i][j]);
    }
    __syncthreads();                                   // reads done before buf reuse
  }

  int n  = lane & 15;
  int mb = hf * 8;
  for (int i = 0; i < 4; ++i)
    for (int j = 0; j < 4; ++j) {
      int r0 = rowBlk + mg * 64 + i * 16 + mb;
      int c0 = col0 + j * 16 + n;
      for (int e = 0; e < 8; ++e)
        OUT[(size_t)(r0 + e) * N + c0] = acc[i][j][e];
    }
}

// ---------------------------------------------------------------------------
// Elementwise fp32 -> bf16 cast
// ---------------------------------------------------------------------------
__global__ void cast_f32_bf16(const float* __restrict__ in, bf16_t* __restrict__ out, int nel) {
  int i = blockIdx.x * blockDim.x + threadIdx.x;
  if (i < nel) out[i] = (bf16_t)in[i];
}

// ---------------------------------------------------------------------------
// RoPE: read fp32 (B*S, D), rotate pairs, write bf16 (B, H, S, HD)
// ---------------------------------------------------------------------------
__global__ void rope_to_bhsd(const float* __restrict__ T, const float* __restrict__ C,
                             const float* __restrict__ Sn, bf16_t* __restrict__ Dst) {
  int idx = blockIdx.x * blockDim.x + threadIdx.x;   // 2*2048*16*64 = 2^22
  if (idx >= (1 << 22)) return;
  int j = idx & 63;
  int h = (idx >> 6) & 15;
  int s = (idx >> 10) & 2047;
  int b = idx >> 21;
  size_t src = (size_t)(b * 2048 + s) * 2048 + h * 128 + 2 * j;
  float re = T[src], im = T[src + 1];
  float c  = C[s * 64 + j], sn = Sn[s * 64 + j];
  float orr = re * c - im * sn;
  float oim = re * sn + im * c;
  size_t dst = ((size_t)(b * 16 + h) * 2048 + s) * 128 + 2 * j;
  Dst[dst]     = (bf16_t)orr;
  Dst[dst + 1] = (bf16_t)oim;
}

// ---------------------------------------------------------------------------
// V transpose: fp32 (B*S, D) -> bf16 (B, H, HD, S)
// ---------------------------------------------------------------------------
__global__ void vt_from_f32(const float* __restrict__ T, bf16_t* __restrict__ Dst) {
  int idx = blockIdx.x * blockDim.x + threadIdx.x;   // 2^23
  if (idx >= (1 << 23)) return;
  int d = idx & 127;
  int h = (idx >> 7) & 15;
  int s = (idx >> 11) & 2047;
  int b = idx >> 22;
  float v = T[(size_t)(b * 2048 + s) * 2048 + h * 128 + d];
  Dst[((size_t)(b * 16 + h) * 128 + d) * 2048 + s] = (bf16_t)v;
}

// ---------------------------------------------------------------------------
// Sliding-window flash attention, one wave per 16-query tile.
// Q,K: (B,H,S,HD) bf16 ; Vt: (B,H,HD,S) bf16 ; O: (B*S, D) bf16
// ---------------------------------------------------------------------------
__global__ __launch_bounds__(128)
void attn_swa(const bf16_t* __restrict__ Q, const bf16_t* __restrict__ Kx,
              const bf16_t* __restrict__ Vt, bf16_t* __restrict__ O) {
  constexpr int SQ = 2048, HDIM = 128, NH = 16, WIN = 256;
  __shared__ __align__(16) bf16_t lds_p[4][16][40];   // 40-el row stride, 16B aligned rows
  int lane = threadIdx.x & 31;
  int wvid = threadIdx.x >> 5;
  int tile = blockIdx.x * 4 + wvid;                   // 0..4095
  int qt = tile & 127;
  int h  = (tile >> 7) & 15;
  int b  = tile >> 11;
  int q0 = qt * 16;
  const bf16_t* Qh = Q  + (size_t)(b * NH + h) * SQ * HDIM;
  const bf16_t* Kh = Kx + (size_t)(b * NH + h) * SQ * HDIM;
  const bf16_t* Vh = Vt + (size_t)(b * NH + h) * HDIM * SQ;

  v16bf qa[4];
  for (int kk = 0; kk < 4; ++kk) qa[kk] = load_a_frag(Qh, HDIM, q0, kk * 32, lane);

  float mrow[8], lrow[8];
  v8f o[8];
  for (int i = 0; i < 8; ++i) { mrow[i] = -1e30f; lrow[i] = 0.0f; }
  for (int f = 0; f < 8; ++f) o[f] = vzero8();

  int n  = lane & 15;
  int hf = lane >> 4;
  int k_lo = q0 - (WIN - 1); if (k_lo < 0) k_lo = 0;
  int k_begin = k_lo & ~31;
  int k_end = q0 + 16;
  const float scale = 0.088388347648318447f;  // 1/sqrt(128)

  for (int kt = k_begin; kt < k_end; kt += 32) {
    v8f s0 = vzero8(), s1 = vzero8();
    for (int kk = 0; kk < 4; ++kk) {
      v16bf kb0 = load_b_frag(Kh, HDIM, kt,      kk * 32, lane);
      v16bf kb1 = load_b_frag(Kh, HDIM, kt + 16, kk * 32, lane);
      s0 = WMMA_BF16(qa[kk], kb0, s0);
      s1 = WMMA_BF16(qa[kk], kb1, s1);
    }
    float p0[8], p1[8], alpha[8];
    for (int i = 0; i < 8; ++i) {
      int iabs = q0 + i + 8 * hf;
      int j0 = kt + n, j1 = kt + 16 + n;
      float v0 = (j0 <= iabs && j0 > iabs - WIN) ? s0[i] * scale : -1e30f;
      float v1 = (j1 <= iabs && j1 > iabs - WIN) ? s1[i] * scale : -1e30f;
      float mx = fmaxf(v0, v1);
      for (int msk = 1; msk < 16; msk <<= 1) mx = fmaxf(mx, __shfl_xor(mx, msk, 32));
      float mnew = fmaxf(mrow[i], mx);
      alpha[i] = __expf(mrow[i] - mnew);
      mrow[i] = mnew;
      v0 = __expf(v0 - mnew);
      v1 = __expf(v1 - mnew);
      p0[i] = v0; p1[i] = v1;
      float rs = v0 + v1;
      for (int msk = 1; msk < 16; msk <<= 1) rs += __shfl_xor(rs, msk, 32);
      lrow[i] = lrow[i] * alpha[i] + rs;
    }
    // Re-layout P (D-fragment) -> A-fragment via LDS (same-wave DS ops are in-order)
    for (int i = 0; i < 8; ++i) {
      int m = i + 8 * hf;
      lds_p[wvid][m][n]      = (bf16_t)p0[i];
      lds_p[wvid][m][16 + n] = (bf16_t)p1[i];
    }
    asm volatile("s_wait_dscnt 0" ::: "memory");
    v16bf pa;
    {
      const bf16_t* lp = &lds_p[wvid][lane & 15][0];
      BF16x16 r;
      r.h[0] = *(const v8bf*)(lp + hf * 8);
      r.h[1] = *(const v8bf*)(lp + 16 + hf * 8);
      pa = r.v;
    }
    for (int f = 0; f < 8; ++f) {
      v8f of = o[f];
      for (int i = 0; i < 8; ++i) of[i] *= alpha[i];
      v16bf vb = load_b_frag(Vh, SQ, f * 16, kt, lane);
      o[f] = WMMA_BF16(pa, vb, of);
    }
  }
  bf16_t* Ob = O + (size_t)b * SQ * (NH * HDIM) + (size_t)h * HDIM;
  for (int f = 0; f < 8; ++f)
    for (int i = 0; i < 8; ++i) {
      int m = i + 8 * hf;
      Ob[(size_t)(q0 + m) * (NH * HDIM) + f * 16 + n] = (bf16_t)(o[f][i] / lrow[i]);
    }
}

// ---------------------------------------------------------------------------
extern "C" void kernel_launch(void* const* d_in, const int* in_sizes, int n_in,
                              void* d_out, int out_size, void* d_ws, size_t ws_size,
                              hipStream_t stream) {
  const float* x  = (const float*)d_in[0];
  const float* fc = (const float*)d_in[1];
  const float* fs = (const float*)d_in[2];
  const float* wq = (const float*)d_in[3];
  const float* wk = (const float*)d_in[4];
  const float* wv = (const float*)d_in[5];
  const float* wo = (const float*)d_in[6];
  float* out = (float*)d_out;
  char* ws = (char*)d_ws;

  const int XEL = 8388608;        // B*S*D
  const int WEL = 4194304;        // D*D
  bf16_t* xb  = (bf16_t*)(ws);                      // 16 MB
  bf16_t* wqb = (bf16_t*)(ws + 16777216);           // 8 MB each
  bf16_t* wkb = wqb + WEL;
  bf16_t* wvb = wkb + WEL;
  bf16_t* wob = wvb + WEL;
  float*  tmp = (float*)(ws + 50331648);            // 32 MB f32 scratch (reused)
  bf16_t* qb  = (bf16_t*)(ws + 83886080);           // (B,H,S,HD)
  bf16_t* kb  = (bf16_t*)(ws + 100663296);          // (B,H,S,HD)
  bf16_t* vt  = (bf16_t*)(ws + 117440512);          // (B,H,HD,S)
  bf16_t* ab  = (bf16_t*)(ws + 134217728);          // (B*S, D)

  cast_f32_bf16<<<(XEL + 255) / 256, 256, 0, stream>>>(x,  xb,  XEL);
  cast_f32_bf16<<<(WEL + 255) / 256, 256, 0, stream>>>(wq, wqb, WEL);
  cast_f32_bf16<<<(WEL + 255) / 256, 256, 0, stream>>>(wk, wkb, WEL);
  cast_f32_bf16<<<(WEL + 255) / 256, 256, 0, stream>>>(wv, wvb, WEL);
  cast_f32_bf16<<<(WEL + 255) / 256, 256, 0, stream>>>(wo, wob, WEL);

  dim3 gg(2048 / 256, 4096 / 128);   // (8, 32)
  gemm_xt_bf16_f32<<<gg, 256, 0, stream>>>(xb, wqb, tmp, 4096, 2048, 2048);
  rope_to_bhsd<<<(1 << 22) / 256, 256, 0, stream>>>(tmp, fc, fs, qb);
  gemm_xt_bf16_f32<<<gg, 256, 0, stream>>>(xb, wkb, tmp, 4096, 2048, 2048);
  rope_to_bhsd<<<(1 << 22) / 256, 256, 0, stream>>>(tmp, fc, fs, kb);
  gemm_xt_bf16_f32<<<gg, 256, 0, stream>>>(xb, wvb, tmp, 4096, 2048, 2048);
  vt_from_f32<<<(1 << 23) / 256, 256, 0, stream>>>(tmp, vt);

  attn_swa<<<4096 / 4, 128, 0, stream>>>(qb, kb, vt, ab);

  gemm_xt_bf16_f32<<<gg, 256, 0, stream>>>(ab, wob, out, 4096, 2048, 2048);
}
